// T5Block_58059367907427
// MI455X (gfx1250) — matrix operand order repaired and checked by reading, not compiled, not run on hardware
//
#include <hip/hip_runtime.h>
#include <hip/hip_bf16.h>
#include <math.h>

// Problem constants (fixed by the reference)
#define BB   4
#define SS   1024
#define SEE  1024
#define DDIM 1024
#define HH   16
#define DKK  64
#define DFFF 4096
#define BSROWS (BB*SS)

typedef __bf16 v16bf __attribute__((ext_vector_type(16)));
typedef __bf16 v8bf  __attribute__((ext_vector_type(8)));
typedef float  v8f   __attribute__((ext_vector_type(8)));

__device__ __forceinline__ unsigned short f2bf(float f) {
  union { float f; unsigned u; } v; v.f = f;
  unsigned r = v.u + 0x7fffu + ((v.u >> 16) & 1u);   // round-to-nearest-even
  return (unsigned short)(r >> 16);
}

__device__ __forceinline__ v16bf mk16(v8bf lo, v8bf hi) {
  v16bf r;
#pragma unroll
  for (int i = 0; i < 8; ++i) { r[i] = lo[i]; r[i + 8] = hi[i]; }
  return r;
}

__device__ __forceinline__ v8f wmma_bf16(v16bf a, v16bf b, v8f c) {
  return __builtin_amdgcn_wmma_f32_16x16x32_bf16(false, a, false, b, (short)0, c,
                                                 false, false);
}

// ---------------------------------------------------------------------------
// Tiled transpose + fp32 -> bf16 convert: dst(N,K) = src(K,N)^T
// ---------------------------------------------------------------------------
__global__ __launch_bounds__(256) void wtrans_kernel(
    const float* __restrict__ src, unsigned short* __restrict__ dst, int K, int N) {
  __shared__ unsigned short tile[32][33];
  const int tx = threadIdx.x & 31;
  const int ty = threadIdx.x >> 5;
  const int k0 = blockIdx.x * 32;
  const int n0 = blockIdx.y * 32;
#pragma unroll
  for (int j = 0; j < 4; ++j)
    tile[ty + j * 8][tx] = f2bf(src[(size_t)(k0 + ty + j * 8) * N + n0 + tx]);
  __syncthreads();
#pragma unroll
  for (int j = 0; j < 4; ++j)
    dst[(size_t)(n0 + ty + j * 8) * K + k0 + tx] = tile[tx][ty + j * 8];
}

// fp32 -> bf16 straight convert
__global__ __launch_bounds__(256) void cvt_kernel(
    const float* __restrict__ src, unsigned short* __restrict__ dst, int n) {
  int i = blockIdx.x * 256 + threadIdx.x;
  if (i < n) dst[i] = f2bf(src[i]);
}

// Per-head V transpose: vt[b,h,dk,s] = v[b,s,h*64+dk], tiled through LDS.
__global__ __launch_bounds__(256) void vtrans_kernel(
    const unsigned short* __restrict__ v, unsigned short* __restrict__ vt, int klen) {
  __shared__ unsigned short tile[32][33];
  const int tx = threadIdx.x & 31;
  const int ty = threadIdx.x >> 5;
  const int s0  = blockIdx.x * 32;
  const int dk0 = blockIdx.y * 32;
  const int bh  = blockIdx.z;                 // b*H + h
  const int b = bh >> 4, h = bh & 15;
#pragma unroll
  for (int j = 0; j < 4; ++j)
    tile[ty + j * 8][tx] =
        v[(size_t)(b * klen + s0 + ty + j * 8) * DDIM + h * DKK + dk0 + tx];
  __syncthreads();
#pragma unroll
  for (int j = 0; j < 4; ++j)
    vt[((size_t)(bh * DKK + dk0 + ty + j * 8)) * klen + s0 + tx] = tile[tx][ty + j * 8];
}

// ---------------------------------------------------------------------------
// T5 RMSNorm: out_bf16[row,:] = w * x * rsqrt(mean(x^2)+eps)
// ---------------------------------------------------------------------------
__global__ __launch_bounds__(256) void rmsnorm_kernel(
    const float* __restrict__ x, const float* __restrict__ w,
    unsigned short* __restrict__ out) {
  __shared__ float red[256];
  const int row = blockIdx.x;
  const float* xr = x + (size_t)row * DDIM;
  float s = 0.f;
#pragma unroll
  for (int i = 0; i < 4; ++i) { float v = xr[threadIdx.x + i * 256]; s += v * v; }
  red[threadIdx.x] = s;
  __syncthreads();
  for (int st = 128; st > 0; st >>= 1) {
    if (threadIdx.x < st) red[threadIdx.x] += red[threadIdx.x + st];
    __syncthreads();
  }
  float scale = rsqrtf(red[0] * (1.0f / DDIM) + 1e-6f);
#pragma unroll
  for (int i = 0; i < 4; ++i) {
    int d = threadIdx.x + i * 256;
    out[(size_t)row * DDIM + d] = f2bf(w[d] * xr[d] * scale);
  }
}

// ---------------------------------------------------------------------------
// T5 relative-position bias LUT: lut[h][rel+1023], rel = key - q.
// ---------------------------------------------------------------------------
__global__ __launch_bounds__(256) void biaslut_kernel(
    const float* __restrict__ table, float* __restrict__ lut) {
  int idx = blockIdx.x * 256 + threadIdx.x;   // over H*2048
  int h = idx >> 11;
  int r = idx & 2047;
  int rel = r - 1023;
  int rb = rel > 0 ? 16 : 0;
  int rp = rel < 0 ? -rel : rel;
  int bucket;
  if (rp < 8) bucket = rp;
  else {
    int large = 8 + (int)(__logf((float)rp * 0.125f) * (8.0f / 2.7725887222397811f));
    bucket = large < 15 ? large : 15;
  }
  lut[idx] = table[(rb + bucket) * HH + h];
}

// ---------------------------------------------------------------------------
// bf16 WMMA GEMM:  C(M,N) = A(M,K) @ Bt(N,K)^T   (Bt holds W^T row-major)
// 256 threads = 8 waves (2x4), block tile 128x128, wave tile 64x32.
// K-loop unrolled x2 with double-buffered fragments for load/WMMA overlap.
// MODE 0: outH = bf16(C); MODE 1: outH = bf16(relu(C)); MODE 2: outF = resid + C
// ---------------------------------------------------------------------------
struct Frags { v16bf a[4]; v16bf b[2]; };

__device__ __forceinline__ void load_frags(Frags& f,
    const unsigned short* __restrict__ A, const unsigned short* __restrict__ Bt,
    int K, int wm, int wn, int l16, int half, int k0) {
#pragma unroll
  for (int mt = 0; mt < 4; ++mt) {
    const unsigned short* p = A + (size_t)(wm + mt * 16 + l16) * K + k0 + half * 16;
    f.a[mt] = mk16(*(const v8bf*)p, *(const v8bf*)(p + 8));
  }
#pragma unroll
  for (int nt = 0; nt < 2; ++nt) {
    const unsigned short* p = Bt + (size_t)(wn + nt * 16 + l16) * K + k0 + half * 16;
    f.b[nt] = *(const v16bf*)p;
  }
}

__device__ __forceinline__ void mma_all(v8f (&acc)[4][2], const Frags& f) {
#pragma unroll
  for (int mt = 0; mt < 4; ++mt)
#pragma unroll
    for (int nt = 0; nt < 2; ++nt)
      acc[mt][nt] = wmma_bf16(f.a[mt], f.b[nt], acc[mt][nt]);
}

template <int MODE>
__global__ __launch_bounds__(256) void gemm_bf16_kernel(
    const unsigned short* __restrict__ A, const unsigned short* __restrict__ Bt,
    int M, int N, int K,
    const float* __restrict__ resid, float* __restrict__ outF,
    unsigned short* __restrict__ outH) {
  const int wave = threadIdx.x >> 5;
  const int lane = threadIdx.x & 31;
  const int l16  = lane & 15;
  const int half = lane >> 4;
  const int wm = blockIdx.y * 128 + (wave >> 2) * 64;
  const int wn = blockIdx.x * 128 + (wave & 3) * 32;

  v8f acc[4][2];
#pragma unroll
  for (int mt = 0; mt < 4; ++mt)
#pragma unroll
    for (int nt = 0; nt < 2; ++nt) acc[mt][nt] = (v8f){0,0,0,0,0,0,0,0};

  Frags f0, f1;
  load_frags(f0, A, Bt, K, wm, wn, l16, half, 0);
  for (int k0 = 0; k0 < K; k0 += 64) {
    load_frags(f1, A, Bt, K, wm, wn, l16, half, k0 + 32);
    if (k0 + 256 < K) {
      // WGP-scope prefetch (locality 3): pull upcoming tiles into WGP$.
#pragma unroll
      for (int mt = 0; mt < 4; ++mt)
        __builtin_prefetch(A + (size_t)(wm + mt * 16 + l16) * K + k0 + 256, 0, 3);
#pragma unroll
      for (int nt = 0; nt < 2; ++nt)
        __builtin_prefetch(Bt + (size_t)(wn + nt * 16 + l16) * K + k0 + 256, 0, 3);
    }
    mma_all(acc, f0);
    if (k0 + 64 < K) load_frags(f0, A, Bt, K, wm, wn, l16, half, k0 + 64);
    mma_all(acc, f1);
  }

#pragma unroll
  for (int mt = 0; mt < 4; ++mt)
#pragma unroll
    for (int nt = 0; nt < 2; ++nt)
#pragma unroll
      for (int v = 0; v < 8; ++v) {
        int row = wm + mt * 16 + v + 8 * half;
        int col = wn + nt * 16 + l16;
        float x = acc[mt][nt][v];
        size_t idx = (size_t)row * N + col;
        if constexpr (MODE == 2) {
          outF[idx] = resid[idx] + x;
        } else {
          if constexpr (MODE == 1) x = fmaxf(x, 0.f);
          outH[idx] = f2bf(x);
        }
      }
}

// ---------------------------------------------------------------------------
// Flash attention (bf16 WMMA, fp32 online softmax), K-frag double buffered.
// Grid: (S/64, H, B); block 128 threads = 4 waves, each wave owns 16 q rows.
// Q,K: (B,len,D) bf16, Vt: (B,H,DK,klen) bf16, Out: (B,S,D) bf16.
// ---------------------------------------------------------------------------
struct KFrag { v16bf t[2][2]; };   // [key-subtile][dk-chunk]

__device__ __forceinline__ void load_kfrag(KFrag& kf,
    const unsigned short* __restrict__ kptr, int kt) {
#pragma unroll
  for (int t = 0; t < 2; ++t) {
    const unsigned short* p = kptr + (size_t)(kt + t * 16) * DDIM;
    kf.t[t][0] = *(const v16bf*)p;          // dk 0..31 (lane half picks 16)
    kf.t[t][1] = *(const v16bf*)(p + 32);   // dk 32..63
  }
}

template <int HAS_BIAS>
__device__ __forceinline__ void attn_step(
    int kt, const KFrag& kf, v16bf qa0, v16bf qa1,
    const unsigned short* __restrict__ vptr, const int* __restrict__ mptr,
    const float* __restrict__ lutq, int l16, int half, int wave, int klen,
    v8f (&o)[4], float (&mrow)[8], float (&lrow)[8],
    unsigned short (*pshare)[16][32]) {
  // V fragments in flight across the score WMMAs + softmax
  v16bf vf[4];
#pragma unroll
  for (int c = 0; c < 4; ++c)
    vf[c] = *(const v16bf*)(vptr + (size_t)(c * 16) * klen + kt);
  const int mk0 = mptr[kt + l16];
  const int mk1 = mptr[kt + 16 + l16];

  // scores: two 16x16 tiles (keys kt..kt+15, kt+16..kt+31)
  v8f s[2];
#pragma unroll
  for (int t = 0; t < 2; ++t) {
    v8f z = (v8f){0,0,0,0,0,0,0,0};
    z = wmma_bf16(qa0, kf.t[t][0], z);
    s[t] = wmma_bf16(qa1, kf.t[t][1], z);
  }

  // scale + bias + mask
#pragma unroll
  for (int t = 0; t < 2; ++t) {
    const int key = kt + t * 16 + l16;
    const int mk  = t ? mk1 : mk0;
#pragma unroll
    for (int v = 0; v < 8; ++v) {
      float sv = s[t][v] * 0.125f;                         // 1/sqrt(64)
      if constexpr (HAS_BIAS) sv += lutq[key - v - 8 * half];
      sv = (mk != 0) ? sv : -1e30f;
      s[t][v] = sv;
    }
  }

  // online softmax (row reductions across the 16-lane half)
  float alpha[8];
#pragma unroll
  for (int v = 0; v < 8; ++v) {
    float mx = fmaxf(s[0][v], s[1][v]);
    mx = fmaxf(mx, __shfl_xor(mx, 1, 32));
    mx = fmaxf(mx, __shfl_xor(mx, 2, 32));
    mx = fmaxf(mx, __shfl_xor(mx, 4, 32));
    mx = fmaxf(mx, __shfl_xor(mx, 8, 32));
    float mnew = fmaxf(mrow[v], mx);
    alpha[v] = __expf(mrow[v] - mnew);
    mrow[v] = mnew;
    float p0 = __expf(s[0][v] - mnew);
    float p1 = __expf(s[1][v] - mnew);
    s[0][v] = p0; s[1][v] = p1;
    float rs = p0 + p1;
    rs += __shfl_xor(rs, 1, 32);
    rs += __shfl_xor(rs, 2, 32);
    rs += __shfl_xor(rs, 4, 32);
    rs += __shfl_xor(rs, 8, 32);
    lrow[v] = lrow[v] * alpha[v] + rs;
  }
#pragma unroll
  for (int c = 0; c < 4; ++c)
#pragma unroll
    for (int v = 0; v < 8; ++v) o[c][v] *= alpha[v];

  // bounce P through per-wave LDS: C-layout -> A-fragment layout
#pragma unroll
  for (int t = 0; t < 2; ++t)
#pragma unroll
    for (int v = 0; v < 8; ++v)
      pshare[wave][v + 8 * half][t * 16 + l16] = f2bf(s[t][v]);
  __builtin_amdgcn_wave_barrier();            // order stores before loads
  v16bf pa = mk16(*(const v8bf*)&pshare[wave][l16][half * 16],
                  *(const v8bf*)&pshare[wave][l16][8 + half * 16]);
  __builtin_amdgcn_wave_barrier();            // order loads before next stores

  // O += P @ V
#pragma unroll
  for (int c = 0; c < 4; ++c) o[c] = wmma_bf16(pa, vf[c], o[c]);
}

template <int HAS_BIAS>
__global__ __launch_bounds__(128) void flash_attn_kernel(
    const unsigned short* __restrict__ Q, const unsigned short* __restrict__ Kb,
    const unsigned short* __restrict__ Vt, const int* __restrict__ mask,
    const float* __restrict__ bias_lut, unsigned short* __restrict__ Out,
    int klen) {
  __shared__ unsigned short pshare[4][16][32];
  const int b    = blockIdx.z;
  const int h    = blockIdx.y;
  const int wave = threadIdx.x >> 5;
  const int lane = threadIdx.x & 31;
  const int l16  = lane & 15;
  const int half = lane >> 4;
  const int q0   = blockIdx.x * 64 + wave * 16;

  const unsigned short* qrow = Q + ((size_t)(b * SS + q0 + l16)) * DDIM + h * DKK;
  v16bf qa0 = mk16(*(const v8bf*)(qrow + half * 16),      *(const v8bf*)(qrow + 8 + half * 16));
  v16bf qa1 = mk16(*(const v8bf*)(qrow + 32 + half * 16), *(const v8bf*)(qrow + 40 + half * 16));

  // loop-invariant per-lane bases (induction-friendly)
  const unsigned short* kptr = Kb + ((size_t)(b * klen + l16)) * DDIM + h * DKK + half * 16;
  const unsigned short* vptr = Vt + ((size_t)((b * HH + h) * DKK + l16)) * klen + half * 16;
  const int*   mptr = mask + b * klen;
  const float* lutq = HAS_BIAS ? (bias_lut + (h << 11) + 1023 - q0) : nullptr;

  v8f o[4];
#pragma unroll
  for (int c = 0; c < 4; ++c) o[c] = (v8f){0,0,0,0,0,0,0,0};
  float mrow[8], lrow[8];
#pragma unroll
  for (int v = 0; v < 8; ++v) { mrow[v] = -3.0e38f; lrow[v] = 0.f; }

  KFrag k0f, k1f;
  load_kfrag(k0f, kptr, 0);
  for (int kt = 0; kt < klen; kt += 64) {       // klen % 64 == 0
    load_kfrag(k1f, kptr, kt + 32);
    attn_step<HAS_BIAS>(kt, k0f, qa0, qa1, vptr, mptr, lutq, l16, half, wave, klen,
                        o, mrow, lrow, pshare);
    if (kt + 64 < klen) load_kfrag(k0f, kptr, kt + 64);
    attn_step<HAS_BIAS>(kt + 32, k1f, qa0, qa1, vptr, mptr, lutq, l16, half, wave, klen,
                        o, mrow, lrow, pshare);
  }

  // normalize + store bf16
  float inv[8];
#pragma unroll
  for (int v = 0; v < 8; ++v) inv[v] = lrow[v] > 0.f ? 1.0f / lrow[v] : 0.f;
#pragma unroll
  for (int c = 0; c < 4; ++c)
#pragma unroll
    for (int v = 0; v < 8; ++v) {
      size_t idx = ((size_t)(b * SS + q0 + v + 8 * half)) * DDIM + h * DKK + c * 16 + l16;
      Out[idx] = f2bf(o[c][v] * inv[v]);
    }
}

// ---------------------------------------------------------------------------
// Host orchestration
// ---------------------------------------------------------------------------
extern "C" void kernel_launch(void* const* d_in, const int* in_sizes, int n_in,
                              void* d_out, int out_size, void* d_ws, size_t ws_size,
                              hipStream_t stream) {
  (void)in_sizes; (void)n_in; (void)out_size; (void)ws_size;
  const float* hidden = (const float*)d_in[0];
  const float* enc    = (const float*)d_in[1];
  const int*   mask1  = (const int*)d_in[2];
  const int*   mask2  = (const int*)d_in[3];
  const float* ln1    = (const float*)d_in[4];
  const float* ln2    = (const float*)d_in[5];
  const float* ln3    = (const float*)d_in[6];
  const float* W[10];
  for (int i = 0; i < 10; ++i) W[i] = (const float*)d_in[7 + i];
  const float* btab   = (const float*)d_in[17];

  char* ws = (char*)d_ws;
  size_t off = 0;
  auto alloc = [&](size_t bytes) -> void* {
    void* p = ws + off;
    off += (bytes + 255) & ~(size_t)255;
    return p;
  };

  int wK[10], wN[10];
  for (int i = 0; i < 8; ++i) { wK[i] = DDIM; wN[i] = DDIM; }
  wK[8] = DDIM;  wN[8] = DFFF;
  wK[9] = DFFF;  wN[9] = DDIM;
  unsigned short* Wt[10];
  for (int i = 0; i < 10; ++i) Wt[i] = (unsigned short*)alloc((size_t)wK[i] * wN[i] * 2);

  unsigned short* ebuf  = (unsigned short*)alloc((size_t)BB * SEE * DDIM * 2);
  unsigned short* nbuf  = (unsigned short*)alloc((size_t)BSROWS * DDIM * 2);
  unsigned short* qbuf  = (unsigned short*)alloc((size_t)BSROWS * DDIM * 2);
  unsigned short* kbuf  = (unsigned short*)alloc((size_t)BSROWS * DDIM * 2);
  unsigned short* vbuf  = (unsigned short*)alloc((size_t)BSROWS * DDIM * 2);
  unsigned short* vtbuf = (unsigned short*)alloc((size_t)BSROWS * DDIM * 2);
  unsigned short* abuf  = (unsigned short*)alloc((size_t)BSROWS * DDIM * 2);
  unsigned short* hbuf  = (unsigned short*)alloc((size_t)BSROWS * DFFF * 2);
  float*          xbuf  = (float*)alloc((size_t)BSROWS * DDIM * 4);
  float*          blut  = (float*)alloc((size_t)HH * 2048 * 4);

  auto gemm_h = [&](const unsigned short* A, const unsigned short* Bt,
                    int M, int N, int K, unsigned short* oH) {
    dim3 g(N / 128, M / 128);
    gemm_bf16_kernel<0><<<g, 256, 0, stream>>>(A, Bt, M, N, K, nullptr, nullptr, oH);
  };
  auto gemm_relu = [&](const unsigned short* A, const unsigned short* Bt,
                       int M, int N, int K, unsigned short* oH) {
    dim3 g(N / 128, M / 128);
    gemm_bf16_kernel<1><<<g, 256, 0, stream>>>(A, Bt, M, N, K, nullptr, nullptr, oH);
  };
  auto gemm_res = [&](const unsigned short* A, const unsigned short* Bt,
                      int M, int N, int K, const float* resid, float* oF) {
    dim3 g(N / 128, M / 128);
    gemm_bf16_kernel<2><<<g, 256, 0, stream>>>(A, Bt, M, N, K, resid, oF, nullptr);
  };

  // weights -> bf16 transposed (tiled), bias LUT, encoder states -> bf16
  for (int i = 0; i < 10; ++i) {
    dim3 g(wK[i] / 32, wN[i] / 32);
    wtrans_kernel<<<g, 256, 0, stream>>>(W[i], Wt[i], wK[i], wN[i]);
  }
  biaslut_kernel<<<(HH * 2048) / 256, 256, 0, stream>>>(btab, blut);
  cvt_kernel<<<(BB * SEE * DDIM) / 256, 256, 0, stream>>>(enc, ebuf, BB * SEE * DDIM);

  // ---- self-attention ----
  rmsnorm_kernel<<<BSROWS, 256, 0, stream>>>(hidden, ln1, nbuf);
  gemm_h(nbuf, Wt[0], BSROWS, DDIM, DDIM, qbuf);
  gemm_h(nbuf, Wt[1], BSROWS, DDIM, DDIM, kbuf);
  gemm_h(nbuf, Wt[2], BSROWS, DDIM, DDIM, vbuf);
  {
    dim3 g(SS / 32, DKK / 32, BB * HH);
    vtrans_kernel<<<g, 256, 0, stream>>>(vbuf, vtbuf, SS);
  }
  {
    dim3 ag(SS / 64, HH, BB);
    flash_attn_kernel<1><<<ag, 128, 0, stream>>>(qbuf, kbuf, vtbuf, mask1, blut, abuf, SS);
  }
  gemm_res(abuf, Wt[3], BSROWS, DDIM, DDIM, hidden, xbuf);   // x = hidden + attn@Wo1

  // ---- cross-attention ----
  rmsnorm_kernel<<<BSROWS, 256, 0, stream>>>(xbuf, ln2, nbuf);
  gemm_h(nbuf, Wt[4], BSROWS, DDIM, DDIM, qbuf);
  gemm_h(ebuf, Wt[5], BB * SEE, DDIM, DDIM, kbuf);
  gemm_h(ebuf, Wt[6], BB * SEE, DDIM, DDIM, vbuf);
  {
    dim3 g(SEE / 32, DKK / 32, BB * HH);
    vtrans_kernel<<<g, 256, 0, stream>>>(vbuf, vtbuf, SEE);
  }
  {
    dim3 ag(SS / 64, HH, BB);
    flash_attn_kernel<0><<<ag, 128, 0, stream>>>(qbuf, kbuf, vtbuf, mask2, nullptr, abuf, SEE);
  }
  gemm_res(abuf, Wt[7], BSROWS, DDIM, DDIM, xbuf, xbuf);     // x += attn@Wo2

  // ---- FFN ----
  rmsnorm_kernel<<<BSROWS, 256, 0, stream>>>(xbuf, ln3, nbuf);
  gemm_relu(nbuf, Wt[8], BSROWS, DFFF, DDIM, hbuf);          // relu(n@W1)
  gemm_res(hbuf, Wt[9], BSROWS, DDIM, DFFF, xbuf, (float*)d_out); // out = x + h@W2
}
